// PausePredictor_17849884082424
// MI455X (gfx1250) — matrix operand order repaired and checked by reading, not compiled, not run on hardware
//
#include <hip/hip_runtime.h>
#include <hip/hip_bf16.h>
#include <stdint.h>

// ---------------------------------------------------------------------------
// PausePredictor for MI455X (gfx1250): bf16 WMMA GEMM formulation of the two
// K=3 conv1d layers, fused flex_sigmoid + LayerNorm epilogues, final linear
// + mask fused into stage2. Weight chunks double-buffered in LDS via the
// CDNA5 async-to-LDS path (ASYNCcnt); intermediates (bf16) live in L2.
// ---------------------------------------------------------------------------

#define BDIM 64
#define TDIM 2048
#define HDIM 256
#define FSD  256
#define KW   3
#define WINW 16
#define XC   288            // conv1 K-per-tap padded 272 -> 288 (9 chunks of 32)
#define XS   296            // LDS row stride (bf16) for stage1 A rows
#define Y1S  264            // LDS row stride (bf16) for stage2 A rows
#define YS   260            // LDS row stride (f32) for conv outputs (bank pad)
#define TT   64             // time rows per workgroup

#define XL1_BYTES 39168     // 66*XS*2 = 39072, rounded to 128
#define XL2_BYTES 34944     // 66*Y1S*2 = 34848, rounded to 128
#define WCHUNK    16384     // one (k,kc) weight chunk: 16 tiles * 1KB
#define SMEM1     (XL1_BYTES + 2 * WCHUNK)   // 71936 (>= 64*YS*4 = 66560)
#define SMEM2     (XL2_BYTES + 2 * WCHUNK)   // 67712 (>= 66560)

typedef __bf16 bf16_t;
typedef __attribute__((ext_vector_type(16))) __bf16        v16bf;
typedef __attribute__((ext_vector_type(8)))  float         v8f;
typedef __attribute__((ext_vector_type(4)))  unsigned int  u32x4;
typedef __attribute__((__vector_size__(4 * sizeof(int)))) int vi4;

union AB { v16bf v; u32x4 q[2]; };

#if defined(__has_builtin)
#if __has_builtin(__builtin_amdgcn_global_load_async_to_lds_b128)
#define HAS_ASYNC_LDS 1
#endif
#if __has_builtin(__builtin_amdgcn_s_wait_asynccnt)
#define HAS_WAIT_ASYNC 1
#endif
#endif

// 16-byte global -> LDS copy. Async (ASYNCcnt-tracked) when available.
__device__ __forceinline__ void copy16_g2l(const void* gp, void* lp) {
#ifdef HAS_ASYNC_LDS
  __builtin_amdgcn_global_load_async_to_lds_b128((vi4*)gp, (vi4*)lp, 0, 0);
#else
  *(u32x4*)lp = *(const u32x4*)gp;
#endif
}

__device__ __forceinline__ void wait_async_all() {
#ifdef HAS_ASYNC_LDS
#ifdef HAS_WAIT_ASYNC
  __builtin_amdgcn_s_wait_asynccnt(0);
#else
  asm volatile("s_wait_asynccnt 0" ::: "memory");
#endif
#endif
}

// Issue one 16KB weight chunk: 256 threads * 64B each.
__device__ __forceinline__ void stage_chunkW(const bf16_t* wsrc,
                                             unsigned char* wdst, int tid) {
  const u32x4* gp = (const u32x4*)wsrc + tid * 4;
  u32x4* lp = (u32x4*)wdst + tid * 4;
#pragma unroll
  for (int s = 0; s < 4; ++s) copy16_g2l(gp + s, lp + s);
}

__device__ __forceinline__ float flexsig(float x, float a, float b, float c,
                                         float invc, float d) {
  float bx  = fabsf(b * x);
  float den = powf(1.0f + powf(bx, c), invc);
  return a * b * x / den + d * x;
}

// ---------------------------------------------------------------------------
// pack_x: Xbf[b][t][0:256]=enc, [256:272]=windowed textEmb (or -1), [272:288]=0
// ---------------------------------------------------------------------------
__global__ void pack_x(const float* __restrict__ enc,
                       const float* __restrict__ temb,
                       bf16_t* __restrict__ xbf) {
  const long long N = (long long)BDIM * TDIM * XC;
  for (long long i = (long long)blockIdx.x * blockDim.x + threadIdx.x; i < N;
       i += (long long)gridDim.x * blockDim.x) {
    int e = (int)(i % XC);
    long long bt = i / XC;
    int t = (int)(bt % TDIM);
    int b = (int)(bt / TDIM);
    float v;
    if (e < HDIM) {
      v = enc[bt * HDIM + e];
    } else if (e < HDIM + WINW) {
      int w = e - HDIM;
      int idx = t - w;
      v = (idx >= 0) ? temb[(long long)b * TDIM + idx] : -1.0f;
    } else {
      v = 0.0f;
    }
    xbf[i] = (bf16_t)v;
  }
}

// ---------------------------------------------------------------------------
// Weight pre-swizzle into WMMA B-operand layout:
//   tile index = (k*NKC + kc)*16 + nt ; per tile: 32 lanes x 16 bf16.
//   lane<16 : column n = nt*16+lane, K rows c = kc*32 + {0..7, 16..23}
//   lane>=16: column n = nt*16+(lane-16), K rows c = kc*32 + {8..15, 24..31}
// ---------------------------------------------------------------------------
__global__ void pack_w1(const float* __restrict__ w1, bf16_t* __restrict__ w1s) {
  const int N = KW * 9 * 16 * 32 * 16;
  for (int i = blockIdx.x * blockDim.x + threadIdx.x; i < N;
       i += gridDim.x * blockDim.x) {
    int e    = i & 15;
    int lane = (i >> 4) & 31;
    int tile = i >> 9;
    int nt = tile & 15;
    int kc = (tile >> 4) % 9;
    int k  = tile / (9 * 16);
    int n  = nt * 16 + (lane & 15);
    int c  = kc * 32 + ((lane >> 4) << 3) + ((e >> 3) << 4) + (e & 7);
    float v = (c < HDIM + WINW)
                  ? w1[((long long)n * (HDIM + WINW) + c) * KW + k]
                  : 0.0f;
    w1s[i] = (bf16_t)v;
  }
}

__global__ void pack_w2(const float* __restrict__ w2, bf16_t* __restrict__ w2s) {
  const int N = KW * 8 * 16 * 32 * 16;
  for (int i = blockIdx.x * blockDim.x + threadIdx.x; i < N;
       i += gridDim.x * blockDim.x) {
    int e    = i & 15;
    int lane = (i >> 4) & 31;
    int tile = i >> 9;
    int nt = tile & 15;
    int kc = (tile >> 4) & 7;
    int k  = tile / (8 * 16);
    int n  = nt * 16 + (lane & 15);
    int c  = kc * 32 + ((lane >> 4) << 3) + ((e >> 3) << 4) + (e & 7);
    w2s[i] = (bf16_t)w2[((long long)n * FSD + c) * KW + k];
  }
}

// ---------------------------------------------------------------------------
// stage1: conv1 (GEMM, 27 K-chunks of 32) + bias + flex_sigmoid + LN -> y1
// Workgroup = 8 wave32; wave (mt,nh) owns 16 rows x 128 cols (8 acc tiles).
// Weight chunks double-buffered in LDS via async-to-LDS; A rows in LDS.
// ---------------------------------------------------------------------------
__global__ __launch_bounds__(256) void stage1(
    const bf16_t* __restrict__ xbf, const bf16_t* __restrict__ w1s,
    const float* __restrict__ b1, const float* __restrict__ fa,
    const float* __restrict__ fb, const float* __restrict__ fc,
    const float* __restrict__ fd, const float* __restrict__ g,
    const float* __restrict__ be, bf16_t* __restrict__ y1) {
  __shared__ __align__(128) unsigned char smem[SMEM1];
  bf16_t* xl = (bf16_t*)smem;
  unsigned char* wbuf = smem + XL1_BYTES;

  const int tid  = threadIdx.x;
  const int lane = tid & 31;
  const int wv   = tid >> 5;
  const int mt   = wv & 3;   // M tile (16 rows each)
  const int nh   = wv >> 2;  // N half (128 cols each)
  const int nb   = TDIM / TT;
  const int bb   = blockIdx.x / nb;
  const int t0   = (blockIdx.x % nb) * TT;
  const u32x4 z4 = {0u, 0u, 0u, 0u};

  // Issue A-row staging: rows t0-1 .. t0+64 (66 rows x 36 x 16B); zero halo.
  for (int j = tid; j < 66 * 36; j += 256) {
    int r = j / 36, c = j % 36;
    int t = t0 - 1 + r;
    u32x4* lrow = (u32x4*)(xl + r * XS);  // 592B row stride (16B aligned)
    if (t >= 0 && t < TDIM)
      copy16_g2l(((const u32x4*)(xbf + ((long long)bb * TDIM + t) * XC)) + c,
                 lrow + c);
    else
      lrow[c] = z4;
  }
  // Issue weight chunk 0 into buffer 0.
  stage_chunkW(w1s, wbuf, tid);

  v8f acc[8];
#pragma unroll
  for (int i = 0; i < 8; ++i) acc[i] = (v8f){0, 0, 0, 0, 0, 0, 0, 0};

  const int arow = mt * 16 + (lane & 15);
  const int asel = (lane >> 4) << 3;  // 0 or 8 (A operand K interleave)

  for (int ch = 0; ch < 27; ++ch) {
    const int p = ch & 1;
    wait_async_all();   // chunk ch (and, first time, A rows) arrived
    __syncthreads();    // published to all waves; buf 1-p free for reuse
    if (ch + 1 < 27)
      stage_chunkW(w1s + (ch + 1) * (WCHUNK / 2), wbuf + (1 - p) * WCHUNK, tid);

    const int k  = ch / 9;
    const int kc = ch - k * 9;
    AB a;
    const bf16_t* ap = xl + (arow + k) * XS + kc * 32 + asel;
    a.q[0] = *(const u32x4*)(ap);
    a.q[1] = *(const u32x4*)(ap + 16);

    const u32x4* bq = (const u32x4*)(wbuf + p * WCHUNK) + nh * 512 + lane * 2;
#pragma unroll
    for (int nt = 0; nt < 8; ++nt) {
      AB bm;
      bm.q[0] = bq[nt * 64];
      bm.q[1] = bq[nt * 64 + 1];
      acc[nt] = __builtin_amdgcn_wmma_f32_16x16x32_bf16(
          false, a.v, false, bm.v, (short)0, acc[nt], false, false);
    }
  }

  __syncthreads();  // done with xl/wbuf; alias as f32 output tile
  float* yt = (float*)smem;
  {
    const int col0 = nh * 128 + (lane & 15);
    const int rb   = mt * 16 + ((lane >> 4) << 3);
#pragma unroll
    for (int nt = 0; nt < 8; ++nt)
#pragma unroll
      for (int r = 0; r < 8; ++r)
        yt[(rb + r) * YS + col0 + nt * 16] = acc[nt][r];
  }
  __syncthreads();

  // Epilogue: 4 lanes per row; strided cols for conflict-free LDS banks.
  const float A = fa[0], Bv = fb[0], C = fc[0], D = fd[0];
  const float invc = 1.0f / C;
  const int r = tid >> 2, q = tid & 3;
  float s = 0.0f, ss = 0.0f;
  for (int i = 0; i < 64; ++i) {
    int c = q + 4 * i;
    float x = yt[r * YS + c] + b1[c];
    x = flexsig(x, A, Bv, C, invc, D);
    yt[r * YS + c] = x;
    s += x;
    ss += x * x;
  }
  s  += __shfl_xor(s, 1);  s  += __shfl_xor(s, 2);
  ss += __shfl_xor(ss, 1); ss += __shfl_xor(ss, 2);
  float mu = s * (1.0f / 256.0f);
  float rs = rsqrtf(ss * (1.0f / 256.0f) - mu * mu + 1e-5f);
  bf16_t* yrow = y1 + ((long long)bb * TDIM + (t0 + r)) * FSD;
  for (int i = 0; i < 64; ++i) {
    int c = q + 4 * i;
    float x = (yt[r * YS + c] - mu) * rs * g[c] + be[c];
    yrow[c] = (bf16_t)x;
  }
}

// ---------------------------------------------------------------------------
// stage2: conv2 (24 K-chunks) + bias + flex_sigmoid + LN + (x@wl + bl) + mask
// ---------------------------------------------------------------------------
__global__ __launch_bounds__(256) void stage2(
    const bf16_t* __restrict__ y1, const bf16_t* __restrict__ w2s,
    const float* __restrict__ b2, const float* __restrict__ fa,
    const float* __restrict__ fb, const float* __restrict__ fc,
    const float* __restrict__ fd, const float* __restrict__ g,
    const float* __restrict__ be, const float* __restrict__ wl,
    const float* __restrict__ bl, const unsigned char* __restrict__ mask,
    float* __restrict__ out) {
  __shared__ __align__(128) unsigned char smem[SMEM2];
  bf16_t* xl = (bf16_t*)smem;
  unsigned char* wbuf = smem + XL2_BYTES;

  const int tid  = threadIdx.x;
  const int lane = tid & 31;
  const int wv   = tid >> 5;
  const int mt   = wv & 3;
  const int nh   = wv >> 2;
  const int nb   = TDIM / TT;
  const int bb   = blockIdx.x / nb;
  const int t0   = (blockIdx.x % nb) * TT;
  const u32x4 z4 = {0u, 0u, 0u, 0u};

  for (int j = tid; j < 66 * 32; j += 256) {
    int r = j / 32, c = j % 32;
    int t = t0 - 1 + r;
    u32x4* lrow = (u32x4*)(xl + r * Y1S);  // 528B row stride (16B aligned)
    if (t >= 0 && t < TDIM)
      copy16_g2l(((const u32x4*)(y1 + ((long long)bb * TDIM + t) * FSD)) + c,
                 lrow + c);
    else
      lrow[c] = z4;
  }
  stage_chunkW(w2s, wbuf, tid);

  v8f acc[8];
#pragma unroll
  for (int i = 0; i < 8; ++i) acc[i] = (v8f){0, 0, 0, 0, 0, 0, 0, 0};

  const int arow = mt * 16 + (lane & 15);
  const int asel = (lane >> 4) << 3;

  for (int ch = 0; ch < 24; ++ch) {
    const int p = ch & 1;
    wait_async_all();
    __syncthreads();
    if (ch + 1 < 24)
      stage_chunkW(w2s + (ch + 1) * (WCHUNK / 2), wbuf + (1 - p) * WCHUNK, tid);

    const int k  = ch >> 3;
    const int kc = ch & 7;
    AB a;
    const bf16_t* ap = xl + (arow + k) * Y1S + kc * 32 + asel;
    a.q[0] = *(const u32x4*)(ap);
    a.q[1] = *(const u32x4*)(ap + 16);

    const u32x4* bq = (const u32x4*)(wbuf + p * WCHUNK) + nh * 512 + lane * 2;
#pragma unroll
    for (int nt = 0; nt < 8; ++nt) {
      AB bm;
      bm.q[0] = bq[nt * 64];
      bm.q[1] = bq[nt * 64 + 1];
      acc[nt] = __builtin_amdgcn_wmma_f32_16x16x32_bf16(
          false, a.v, false, bm.v, (short)0, acc[nt], false, false);
    }
  }

  __syncthreads();
  float* yt = (float*)smem;
  {
    const int col0 = nh * 128 + (lane & 15);
    const int rb   = mt * 16 + ((lane >> 4) << 3);
#pragma unroll
    for (int nt = 0; nt < 8; ++nt)
#pragma unroll
      for (int r = 0; r < 8; ++r)
        yt[(rb + r) * YS + col0 + nt * 16] = acc[nt][r];
  }
  __syncthreads();

  const float A = fa[0], Bv = fb[0], C = fc[0], D = fd[0];
  const float invc = 1.0f / C;
  const int r = tid >> 2, q = tid & 3;
  float s = 0.0f, ss = 0.0f;
  for (int i = 0; i < 64; ++i) {
    int c = q + 4 * i;
    float x = yt[r * YS + c] + b2[c];
    x = flexsig(x, A, Bv, C, invc, D);
    yt[r * YS + c] = x;
    s += x;
    ss += x * x;
  }
  s  += __shfl_xor(s, 1);  s  += __shfl_xor(s, 2);
  ss += __shfl_xor(ss, 1); ss += __shfl_xor(ss, 2);
  float mu = s * (1.0f / 256.0f);
  float rs = rsqrtf(ss * (1.0f / 256.0f) - mu * mu + 1e-5f);
  float dot = 0.0f;
  for (int i = 0; i < 64; ++i) {
    int c = q + 4 * i;
    float x = (yt[r * YS + c] - mu) * rs * g[c] + be[c];
    dot += x * wl[c];
  }
  dot += __shfl_xor(dot, 1);
  dot += __shfl_xor(dot, 2);
  if (q == 0) {
    long long idx = (long long)bb * TDIM + (t0 + r);
    out[idx] = mask[idx] ? 0.0f : (dot + bl[0]);
  }
}

// ---------------------------------------------------------------------------
extern "C" void kernel_launch(void* const* d_in, const int* in_sizes, int n_in,
                              void* d_out, int out_size, void* d_ws,
                              size_t ws_size, hipStream_t stream) {
  (void)in_sizes; (void)n_in; (void)out_size; (void)ws_size;
  const float* enc  = (const float*)d_in[0];
  const unsigned char* mask = (const unsigned char*)d_in[1];
  const float* temb = (const float*)d_in[2];
  const float* w1   = (const float*)d_in[3];
  const float* b1   = (const float*)d_in[4];
  const float* fa1  = (const float*)d_in[5];
  const float* fb1  = (const float*)d_in[6];
  const float* fc1  = (const float*)d_in[7];
  const float* fd1  = (const float*)d_in[8];
  const float* g1   = (const float*)d_in[9];
  const float* be1  = (const float*)d_in[10];
  const float* w2   = (const float*)d_in[11];
  const float* b2   = (const float*)d_in[12];
  const float* fa2  = (const float*)d_in[13];
  const float* fb2  = (const float*)d_in[14];
  const float* fc2  = (const float*)d_in[15];
  const float* fd2  = (const float*)d_in[16];
  const float* g2   = (const float*)d_in[17];
  const float* be2  = (const float*)d_in[18];
  const float* wl   = (const float*)d_in[19];
  const float* bl   = (const float*)d_in[20];
  float* out = (float*)d_out;

  unsigned char* ws = (unsigned char*)d_ws;
  size_t o = 0;
  bf16_t* xbf = (bf16_t*)(ws + o); o += (size_t)BDIM * TDIM * XC * 2;      // 75.5 MB
  bf16_t* w1s = (bf16_t*)(ws + o); o += (size_t)KW * 9 * 16 * 32 * 16 * 2; // 432 KB
  bf16_t* w2s = (bf16_t*)(ws + o); o += (size_t)KW * 8 * 16 * 32 * 16 * 2; // 384 KB
  bf16_t* y1  = (bf16_t*)(ws + o);                                         // 64 MB

  pack_x<<<8192, 256, 0, stream>>>(enc, temb, xbf);
  pack_w1<<<432, 256, 0, stream>>>(w1, w1s);
  pack_w2<<<384, 256, 0, stream>>>(w2, w2s);

  const int nblk = BDIM * (TDIM / TT);  // 2048
  stage1<<<nblk, 256, 0, stream>>>(xbf, w1s, b1, fa1, fb1, fc1, fd1, g1, be1, y1);
  stage2<<<nblk, 256, 0, stream>>>(y1, w2s, b2, fa2, fb2, fc2, fd2, g2, be2, wl,
                                   bl, mask, out);
}
